// Reservoir_29781303230624
// MI455X (gfx1250) — compile-verified, matrix operand order checked by reading, not run on hardware
//
#include <hip/hip_runtime.h>
#include <hip/hip_bf16.h>
#include <math.h>

// Problem constants (match reference)
#define BB   32
#define TT   512
#define DD   256
#define RR   2048
#define KTOT 2304          // R + D combined K dimension
#define LDSSTR 2312        // 2304 + 8 bf16 pad (16B) to rotate LDS banks per row
#define NBLK 32            // persistent grid size (must all be co-resident)

typedef __attribute__((ext_vector_type(16))) __bf16 bf16x16;
typedef __attribute__((ext_vector_type(8)))  __bf16 bf16x8;
typedef __attribute__((ext_vector_type(8)))  float  floatx8;

// ---------------------------------------------------------------------------
// Kernel 1: per-sequence lengths (count timesteps whose feature-sum == 0)
// ---------------------------------------------------------------------------
__global__ void esn_lengths_kernel(const float* __restrict__ batch,
                                   int* __restrict__ len,
                                   float* __restrict__ out_len) {
    int b = blockIdx.x;                       // 32 blocks, 256 threads
    __shared__ int zc;
    if (threadIdx.x == 0) zc = 0;
    __syncthreads();
    for (int t = threadIdx.x; t < TT; t += blockDim.x) {
        const float* row = batch + ((size_t)b * TT + t) * DD;
        float s = 0.0f;
        for (int d = 0; d < DD; ++d) s += row[d];
        if (s == 0.0f) atomicAdd(&zc, 1);
    }
    __syncthreads();
    if (threadIdx.x == 0) {
        int L = TT - zc;
        len[b] = L;
        out_len[b] = (float)L;                // lengths appended to output
    }
}

// ---------------------------------------------------------------------------
// Kernel 2: build combined bf16 weight matrix Wc[r][k] = [Wres | Win] and
//           broadcast initial state into the f32 ping buffer.
// ---------------------------------------------------------------------------
__global__ void esn_prep_kernel(const float* __restrict__ win,
                                const float* __restrict__ wres,
                                const float* __restrict__ init,
                                __bf16* __restrict__ Wc,
                                float* __restrict__ X0) {
    size_t i0 = (size_t)blockIdx.x * blockDim.x + threadIdx.x;
    size_t stride = (size_t)gridDim.x * blockDim.x;
    const size_t NW = (size_t)RR * KTOT;
    for (size_t i = i0; i < NW; i += stride) {
        size_t r = i / KTOT, k = i % KTOT;
        float v = (k < RR) ? wres[r * RR + k] : win[r * DD + (k - RR)];
        Wc[i] = (__bf16)v;
    }
    const size_t NX = (size_t)BB * RR;
    for (size_t i = i0; i < NX; i += stride) {
        X0[i] = init[i % RR];
    }
}

// ---------------------------------------------------------------------------
// Device-wide two-phase barrier (counter + generation in global workspace).
// ---------------------------------------------------------------------------
__device__ __forceinline__ void grid_sync(unsigned* cnt, unsigned* gen) {
    __syncthreads();
    if (threadIdx.x == 0) {
        __threadfence();
        unsigned g = __hip_atomic_load(gen, __ATOMIC_RELAXED, __HIP_MEMORY_SCOPE_AGENT);
        unsigned prev = __hip_atomic_fetch_add(cnt, 1u, __ATOMIC_ACQ_REL, __HIP_MEMORY_SCOPE_AGENT);
        if (prev == NBLK - 1) {
            __hip_atomic_store(cnt, 0u, __ATOMIC_RELAXED, __HIP_MEMORY_SCOPE_AGENT);
            __hip_atomic_fetch_add(gen, 1u, __ATOMIC_RELEASE, __HIP_MEMORY_SCOPE_AGENT);
        } else {
            while (__hip_atomic_load(gen, __ATOMIC_ACQUIRE, __HIP_MEMORY_SCOPE_AGENT) == g) {
                __builtin_amdgcn_s_sleep(1);
            }
        }
        __threadfence();
    }
    __syncthreads();
}

// ---------------------------------------------------------------------------
// Kernel 3: persistent reservoir scan.
//   32 blocks x 256 threads (8 wave32 per block); one 16x16 output tile per
//   wave for the whole scan, so the wave carries its 8 state elements in
//   registers across timesteps (no state re-load in the epilogue).
//   Per step: 72 v_wmma_f32_16x16x32_bf16 over K=2304 on TWO independent
//   accumulator chains (ILP), then tanh/leak/mask epilogue, ping-pong state
//   buffer (for the A-operand staging of all blocks), device-wide barrier.
// ---------------------------------------------------------------------------
__global__ __launch_bounds__(256, 1)
void esn_scan_kernel(const float* __restrict__ batch,
                     const __bf16* __restrict__ Wc,
                     const float* __restrict__ bias,
                     const int* __restrict__ len,
                     const float* __restrict__ init,
                     float* __restrict__ Xbase,   // [2][BB][RR] ping-pong
                     float* __restrict__ out,
                     unsigned* cnt, unsigned* gen) {
    // 16 rows of [x | inp_t] in bf16, padded row stride
    __shared__ __align__(16) __bf16 Xs[16 * LDSSTR];

    const int tid  = threadIdx.x;
    const int bid  = blockIdx.x;
    const int wid  = tid >> 5;
    const int lane = tid & 31;

    const int tile = bid * 8 + wid;           // 0..255
    const int mt   = tile >> 7;               // M-tile: 0 or 1 (uniform per block)
    const int nb   = (tile & 127) * 16;       // N base (reservoir unit)
    const int mbase = mt * 16;                // batch-row base for this block

    const int lm = lane & 15;                 // A/B fragment row/col within tile
    const int kb = (lane >> 4) * 8;           // K half-select per ISA layout

    // ---- per-scan invariants: C/D element -> (b, r) mapping is static ----
    const int n = lane & 15;                  // C/D layout: N = lane&15
    const int r = nb + n;
    const int moff = (lane >> 4) * 8;         // M = vgpr + 8 for upper lane half
    const float bs = bias[r];
    int   lenv[8];
    float xprev[8];
    #pragma unroll
    for (int e = 0; e < 8; ++e) {
        lenv[e]  = len[mbase + e + moff];
        xprev[e] = init[r];                   // x0 broadcast across batch
    }

    for (int t = 0; t < TT; ++t) {
        // single base pointer + uniform offset keeps loads in global addrspace
        const float* Xc = Xbase + (size_t)(t & 1) * (BB * RR);
        float*       Xn = Xbase + (size_t)((t + 1) & 1) * (BB * RR);

        if (t) grid_sync(cnt, gen);           // prev-step state visible device-wide

        // ---- stage A operand: 16 rows x 2304 cols, f32 -> bf16 into LDS ----
        const int CHUNKS = (16 * KTOT) / 4;   // float4-sized chunks
        for (int c = tid; c < CHUNKS; c += 256) {
            int idx = c * 4;
            int row = idx / KTOT;
            int col = idx - row * KTOT;       // KTOT divisible by 4: chunk stays in row
            int b = mbase + row;
            float4 v;
            if (col < RR) {
                v = *(const float4*)&Xc[(size_t)b * RR + col];
            } else {
                v = *(const float4*)&batch[((size_t)b * TT + t) * DD + (col - RR)];
            }
            __bf16* dst = &Xs[row * LDSSTR + col];
            dst[0] = (__bf16)v.x; dst[1] = (__bf16)v.y;
            dst[2] = (__bf16)v.z; dst[3] = (__bf16)v.w;
        }
        __syncthreads();

        // ---- K loop: 72 x v_wmma_f32_16x16x32_bf16, two independent chains --
        floatx8 acc0 = {};
        floatx8 acc1 = {};
        for (int kc = 0; kc < KTOT; kc += 64) {
            #pragma unroll
            for (int u = 0; u < 2; ++u) {
                const int k0 = kc + u * 32;
                // A fragment (ISA 16-bit A 16x32 layout): lane lm = row M,
                // K = {k0+kb .. +7} and {k0+16+kb .. +7}
                const __bf16* arow = &Xs[lm * LDSSTR + k0 + kb];
                bf16x8 alo = *(const bf16x8*)arow;          // ds_load_b128
                bf16x8 ahi = *(const bf16x8*)(arow + 16);
                // B fragment: column N = lm of W^T == row (nb+lm) of Wc
                const __bf16* brow = &Wc[(size_t)(nb + lm) * KTOT + k0 + kb];
                bf16x8 blo = *(const bf16x8*)brow;          // global_load_b128 (L2-hot)
                bf16x8 bhi = *(const bf16x8*)(brow + 16);

                bf16x16 a, b;
                #pragma unroll
                for (int j = 0; j < 8; ++j) {
                    a[j] = alo[j]; a[8 + j] = ahi[j];
                    b[j] = blo[j]; b[8 + j] = bhi[j];
                }
                if (u == 0)
                    acc0 = __builtin_amdgcn_wmma_f32_16x16x32_bf16(
                        false, a, false, b, (short)0, acc0, false, false);
                else
                    acc1 = __builtin_amdgcn_wmma_f32_16x16x32_bf16(
                        false, a, false, b, (short)0, acc1, false, false);
            }
        }

        // ---- epilogue: tanh, leak (0.5/0.5), mask; state carried in regs ----
        #pragma unroll
        for (int e = 0; e < 8; ++e) {
            int b = mbase + e + moff;
            float pre = acc0[e] + acc1[e] + bs;
            float xn  = 0.5f * xprev[e] + 0.5f * tanhf(pre);
            xn = (t < lenv[e]) ? xn : 0.0f;
            xprev[e] = xn;
            Xn[(size_t)b * RR + r] = xn;              // for all blocks' A staging
            out[((size_t)b * TT + t) * RR + r] = xn;  // result tensor
        }
    }
}

// ---------------------------------------------------------------------------
extern "C" void kernel_launch(void* const* d_in, const int* in_sizes, int n_in,
                              void* d_out, int out_size, void* d_ws, size_t ws_size,
                              hipStream_t stream) {
    const float* batch       = (const float*)d_in[0];   // (B,T,D)
    const float* input_w     = (const float*)d_in[1];   // (R,D)
    const float* reservoir_w = (const float*)d_in[2];   // (R,R)
    const float* bias        = (const float*)d_in[3];   // (R,)
    const float* init_state  = (const float*)d_in[4];   // (R,)
    float* out = (float*)d_out;                         // (B,T,R) states ++ (B,) lengths

    // workspace layout
    char* ws = (char*)d_ws;
    unsigned* sync = (unsigned*)ws;                     // [cnt, gen]
    int* len       = (int*)(ws + 64);                   // 32 ints
    float* Xbase   = (float*)(ws + 256);                // 2 x 32x2048 f32 ping-pong
    __bf16* Wc     = (__bf16*)(ws + 256 + 2u * BB * RR * sizeof(float)); // 2048x2304 bf16

    // barrier state must start at 0 (workspace is poisoned); graph-capturable
    hipMemsetAsync(sync, 0, 2 * sizeof(unsigned), stream);

    esn_lengths_kernel<<<BB, 256, 0, stream>>>(
        batch, len, out + (size_t)BB * TT * RR);

    esn_prep_kernel<<<1184, 256, 0, stream>>>(
        input_w, reservoir_w, init_state, Wc, Xbase);

    esn_scan_kernel<<<NBLK, 256, 0, stream>>>(
        batch, Wc, bias, len, init_state, Xbase, out, sync, sync + 1);
}
